// CompleteMOERTDETR_78606491451928
// MI455X (gfx1250) — matrix operand order repaired and checked by reading, not compiled, not run on hardware
//
#include <hip/hip_runtime.h>

// MoE RT-DETR routing + combine for MI455X (gfx1250, wave32, WMMA).
//
// Roofline: GEMM1 = 38400 x 128 x 256 (~2.5 GFLOP), total memory ~50 MB
// -> memory floor ~2.2us @ 23.3 TB/s. f16 WMMA makes compute negligible.

typedef _Float16 v16h __attribute__((ext_vector_type(16)));
typedef _Float16 v8h  __attribute__((ext_vector_type(8)));
typedef float    v8f  __attribute__((ext_vector_type(8)));

#define B_   128
#define S_   400
#define D_   256
#define DH   128          // D/2
#define E_   6
#define Q_   300
#define NTOK (B_ * Q_)    // 38400 tokens actually used (n = min(S,Q) = 300)

#define W1T_STRIDE 264    // halves per W1^T row (256 + 8 pad -> conflict-free ds_read_b128)
#define HS_STRIDE  132    // floats per H row (128 + 4 pad)

// ---------------------------------------------------------------------------
// Kernel 1: token routing.
//  - GEMM1 (WMMA f16, f32 accum): H = relu(X @ W1 + b1), X = 128-row tile
//  - GEMM2 (VALU, tiny N=6): logits = H @ W2 + b2
//  - softmax -> top2 -> softmax(top2) -> 16B record per token in workspace
// ---------------------------------------------------------------------------
__global__ __launch_bounds__(256)
void moe_route_kernel(const float* __restrict__ features,
                      const float* __restrict__ W1,
                      const float* __restrict__ b1,
                      const float* __restrict__ W2,
                      const float* __restrict__ b2,
                      float4* __restrict__ route)
{
    // W1^T staged as f16 (padded) + H tile (padded), separate buffers.
    __shared__ __align__(16) _Float16 w1t[DH * W1T_STRIDE];   // ~67.6 KB
    __shared__ __align__(16) float    Hs[128 * HS_STRIDE];    // ~67.6 KB

    const int tid  = threadIdx.x;
    const int lane = tid & 31;
    const int wave = tid >> 5;
    const int m0   = blockIdx.x * 128;       // first token row of this block

    // ---- Stage W1^T into LDS as f16: w1t[n][k] = (half)W1[k*DH + n] ----
    for (int i = tid; i < D_ * DH; i += 256) {
        int k = i >> 7;          // 0..255
        int n = i & 127;         // 0..127
        w1t[n * W1T_STRIDE + k] = (_Float16)W1[k * DH + n];
    }
    __syncthreads();

    // ---- GEMM1 via v_wmma_f32_16x16x32_f16 ----
    // A layout (16-bit A 16x32): lane&15 = M; lanes<16 hold K {0..7,16..23},
    // lanes>=16 hold K {8..15,24..31}.
    // B layout (16-bit B 32x16): lane&15 = N; lanes<16 hold K {0..7,16..23},
    // lanes>=16 hold K {8..15,24..31}.
    const int nlo   = lane & 15;
    const int koffA = (lane < 16) ? 0 : 8;
    const int koffB = (lane < 16) ? 0 : 8;

    const int rowt = m0 + wave * 16 + nlo;   // token this lane's A rows map to
    const int bb   = rowt / Q_;
    const int ss   = rowt - bb * Q_;
    const float* arow = features + (size_t)(bb * S_ + ss) * D_;

    v8f acc[8] = {};   // 8 N-tiles of 16 -> full N=128 strip, 16 rows

    for (int kc = 0; kc < D_; kc += 32) {
        // A fragment: two 8-float runs per lane, converted to f16.
        float4 a0 = *(const float4*)(arow + kc + koffA);
        float4 a1 = *(const float4*)(arow + kc + koffA + 4);
        float4 a2 = *(const float4*)(arow + kc + 16 + koffA);
        float4 a3 = *(const float4*)(arow + kc + 16 + koffA + 4);
        v16h A;
        A[0]  = (_Float16)a0.x; A[1]  = (_Float16)a0.y;
        A[2]  = (_Float16)a0.z; A[3]  = (_Float16)a0.w;
        A[4]  = (_Float16)a1.x; A[5]  = (_Float16)a1.y;
        A[6]  = (_Float16)a1.z; A[7]  = (_Float16)a1.w;
        A[8]  = (_Float16)a2.x; A[9]  = (_Float16)a2.y;
        A[10] = (_Float16)a2.z; A[11] = (_Float16)a2.w;
        A[12] = (_Float16)a3.x; A[13] = (_Float16)a3.y;
        A[14] = (_Float16)a3.z; A[15] = (_Float16)a3.w;

        #pragma unroll
        for (int nt = 0; nt < 8; ++nt) {
            const _Float16* bp = &w1t[(nt * 16 + nlo) * W1T_STRIDE + kc + koffB];
            v8h blo = *(const v8h*)bp;        // ds_read_b128
            v8h bhi = *(const v8h*)(bp + 16); // ds_read_b128
            v16h Bf;
            #pragma unroll
            for (int i = 0; i < 8; ++i) { Bf[i] = blo[i]; Bf[i + 8] = bhi[i]; }
            acc[nt] = __builtin_amdgcn_wmma_f32_16x16x32_f16(
                false, A, false, Bf, (short)0, acc[nt], false, false);
        }
    }

    // ---- bias + relu, scatter C layout (VGPR r: M=r / M=8+r) into H tile ----
    const int mhi = (lane < 16) ? 0 : 8;
    #pragma unroll
    for (int nt = 0; nt < 8; ++nt) {
        int n = nt * 16 + nlo;
        float bias = b1[n];
        #pragma unroll
        for (int r = 0; r < 8; ++r) {
            int m = wave * 16 + mhi + r;
            float v = acc[nt][r] + bias;
            Hs[m * HS_STRIDE + n] = v > 0.f ? v : 0.f;
        }
    }
    __syncthreads();

    // ---- GEMM2 + softmax + top2 + routing softmax (threads 0..127, 1 row each)
    if (tid < 128) {
        float l[E_];
        #pragma unroll
        for (int e = 0; e < E_; ++e) l[e] = b2[e];
        const float* hrow = &Hs[tid * HS_STRIDE];
        for (int k = 0; k < DH; ++k) {
            float h = hrow[k];
            #pragma unroll
            for (int e = 0; e < E_; ++e) l[e] += h * W2[k * E_ + e];
        }
        // softmax over 6 experts
        float mx = l[0];
        #pragma unroll
        for (int e = 1; e < E_; ++e) mx = fmaxf(mx, l[e]);
        float sum = 0.f, p[E_];
        #pragma unroll
        for (int e = 0; e < E_; ++e) { p[e] = __expf(l[e] - mx); sum += p[e]; }
        float inv = 1.f / sum;
        // top-2 (lowest index wins ties, matching jax.lax.top_k)
        int i0 = 0; float v0 = p[0] * inv;
        #pragma unroll
        for (int e = 1; e < E_; ++e) {
            float pe = p[e] * inv;
            if (pe > v0) { v0 = pe; i0 = e; }
        }
        int i1 = -1; float v1 = -1.f;
        #pragma unroll
        for (int e = 0; e < E_; ++e) {
            if (e == i0) continue;
            float pe = p[e] * inv;
            if (pe > v1) { v1 = pe; i1 = e; }
        }
        // routing = softmax([v0, v1]); v0 >= v1 so this is numerically safe
        float w0 = 1.f / (1.f + __expf(v1 - v0));
        float4 rec;
        rec.x = __int_as_float(i0);
        rec.y = __int_as_float(i1);
        rec.z = w0;
        rec.w = 1.f - w0;
        route[m0 + tid] = rec;
    }
}

// ---------------------------------------------------------------------------
// Kernel 2: gather the two selected experts' predictions and blend.
// Pure streaming: ~10 MB total traffic.
// ---------------------------------------------------------------------------
__global__ __launch_bounds__(256)
void moe_combine_kernel(const float4* __restrict__ route,
                        const float* __restrict__ eb,   // (E,B,Q,4)
                        const float* __restrict__ ec,   // (E,B,Q,6)
                        float* __restrict__ out_bbox,   // (B,Q,4)
                        float* __restrict__ out_cls)    // (B,Q,6)
{
    int t = blockIdx.x * 256 + threadIdx.x;
    if (t >= NTOK) return;

    float4 rec = route[t];
    int   e0 = __float_as_int(rec.x);
    int   e1 = __float_as_int(rec.y);
    float w0 = rec.z, w1 = rec.w;

    size_t bq  = (size_t)t;          // t == b*Q_ + q already
    size_t bq4 = bq * 4;
    size_t bq6 = bq * 6;

    const float4* p0 = (const float4*)(eb + (size_t)e0 * NTOK * 4 + bq4);
    const float4* p1 = (const float4*)(eb + (size_t)e1 * NTOK * 4 + bq4);
    float4 x0 = *p0, x1 = *p1;
    float4 o;
    o.x = w0 * x0.x + w1 * x1.x;
    o.y = w0 * x0.y + w1 * x1.y;
    o.z = w0 * x0.z + w1 * x1.z;
    o.w = w0 * x0.w + w1 * x1.w;
    *(float4*)(out_bbox + bq4) = o;

    const float* c0 = ec + (size_t)e0 * NTOK * 6 + bq6;
    const float* c1 = ec + (size_t)e1 * NTOK * 6 + bq6;
    #pragma unroll
    for (int c = 0; c < 6; ++c)
        out_cls[bq6 + c] = w0 * c0[c] + w1 * c1[c];
}

// ---------------------------------------------------------------------------
extern "C" void kernel_launch(void* const* d_in, const int* in_sizes, int n_in,
                              void* d_out, int out_size, void* d_ws, size_t ws_size,
                              hipStream_t stream)
{
    const float* features = (const float*)d_in[0];  // (128,400,256)
    const float* W1       = (const float*)d_in[1];  // (256,128)
    const float* b1       = (const float*)d_in[2];  // (128)
    const float* W2       = (const float*)d_in[3];  // (128,6)
    const float* b2       = (const float*)d_in[4];  // (6)
    const float* eb       = (const float*)d_in[5];  // (6,128,300,4)
    const float* ec       = (const float*)d_in[6];  // (6,128,300,6)

    float* out_bbox = (float*)d_out;                       // 128*300*4 floats
    float* out_cls  = (float*)d_out + (size_t)NTOK * 4;    // 128*300*6 floats

    float4* route = (float4*)d_ws;   // NTOK * 16B = 600 KB scratch

    // 300 blocks x 128 tokens = 38400 = all tokens with s < n (=300).
    moe_route_kernel<<<NTOK / 128, 256, 0, stream>>>(
        features, W1, b1, W2, b2, route);

    moe_combine_kernel<<<(NTOK + 255) / 256, 256, 0, stream>>>(
        route, eb, ec, out_bbox, out_cls);
}
